// AttentionLayer_72481868087513
// MI455X (gfx1250) — matrix operand order repaired and checked by reading, not compiled, not run on hardware
//
#include <hip/hip_runtime.h>

// MI455X (gfx1250, wave32) track-attention:
//   K1: QKV projection GEMM (f32 in -> f16 A/B, f32 acc WMMA), scatter to
//       Q[T][H][L][Dh], Kt[T][H][Dh][L] (transposed!), V[T][H][L][Dh] in f16.
//   K2: flash attention per (track, head): Q,Kt,V staged in LDS via
//       global_load_async_to_lds_b128 (ASYNCcnt path), scores via
//       v_wmma_f32_16x16x32_f16, online softmax, P*V via WMMA.
//   K3: output projection GEMM -> f32 d_out.
// Sizes hardcoded to the reference setup: N=65536, d=256, H=8, T=256, L=256, Dh=32.
// Workspace: Q(32MB) + Kt(32MB) + V(32MB) + ctx(64MB) = 160MB.

typedef __attribute__((ext_vector_type(16))) _Float16 v16h;
typedef __attribute__((ext_vector_type(8)))  float    v8f;
typedef int v4i_ __attribute__((vector_size(16)));

union V16U { uint4 u[2]; v16h h; };

static __device__ __forceinline__ v8f vzero8() {
  v8f z;
#pragma unroll
  for (int i = 0; i < 8; ++i) z[i] = 0.0f;
  return z;
}

#define WMMA_F16(a, b, c) \
  __builtin_amdgcn_wmma_f32_16x16x32_f16(false, (a), false, (b), (short)0, (c), false, false)

static __device__ __forceinline__ void lds_wave_fence() {
  // Within-wave LDS RAW is handled in-order by hardware; this is a compiler
  // ordering barrier + explicit dscnt drain for safety.
  __asm__ volatile("" ::: "memory");
#if __has_builtin(__builtin_amdgcn_s_wait_dscnt)
  __builtin_amdgcn_s_wait_dscnt(0);
#else
  __asm__ volatile("s_wait_dscnt 0" ::: "memory");
#endif
  __asm__ volatile("" ::: "memory");
}

// ---- Async global->LDS (CDNA5 GLOBAL_LOAD_ASYNC_TO_LDS_B128, ASYNCcnt) ----
// Probe-derived signature: (v4i addrspace(1)*, v4i addrspace(3)*, imm, imm).
#if __has_builtin(__builtin_amdgcn_global_load_async_to_lds_b128)
#define HAVE_ASYNC_LDS 1
static __device__ __forceinline__ void async_copy16(const void* g, void* l) {
  __builtin_amdgcn_global_load_async_to_lds_b128(
      (__attribute__((address_space(1))) v4i_*)g,
      (__attribute__((address_space(3))) v4i_*)l, 0, 0);
}
#endif

static __device__ __forceinline__ void wait_asynccnt0() {
#if __has_builtin(__builtin_amdgcn_s_wait_asynccnt)
  __builtin_amdgcn_s_wait_asynccnt(0);
#else
  __asm__ volatile("s_wait_asynccnt 0" ::: "memory");
#endif
}

// ---------------------------------------------------------------------------
// Kernel 1: qkv = values @ w_qkv^T + b_qkv, scattered into per-track layout.
// Block: 128 rows x 64 cols. Grid: (512, 12).
// ---------------------------------------------------------------------------
__global__ __launch_bounds__(256) void qkv_gemm_kernel(
    const float* __restrict__ values, const float* __restrict__ w_qkv,
    const float* __restrict__ b_qkv, const int* __restrict__ track_ids,
    _Float16* __restrict__ qbuf, _Float16* __restrict__ ktbuf,
    _Float16* __restrict__ vbuf) {
  constexpr int K = 256, L = 256, H = 8, Dh = 32;
  __shared__ __align__(16) _Float16 As[128][40];  // 128 rows x 32 k (pad 8)
  __shared__ __align__(16) _Float16 Bs[32][72];   // 32 k x 64 cols (pad 8)

  const int tid  = threadIdx.x;
  const int lane = tid & 31, wave = tid >> 5;
  const int half = lane >> 4, l16 = lane & 15;
  const int rowBase = blockIdx.x * 128;
  const int colBase = blockIdx.y * 64;

  v8f acc[4];
#pragma unroll
  for (int ti = 0; ti < 4; ++ti) acc[ti] = vzero8();

  for (int kb = 0; kb < K; kb += 32) {
    // Stage A tile (128x32) as f16.
#pragma unroll
    for (int e = 0; e < 16; ++e) {
      int idx = e * 256 + tid;
      int r = idx >> 5, kk = idx & 31;
      As[r][kk] = (_Float16)values[(size_t)(rowBase + r) * K + kb + kk];
    }
    // Stage B tile transposed: Bs[kk][c] = w_qkv[colBase+c][kb+kk].
#pragma unroll
    for (int e = 0; e < 8; ++e) {
      int idx = e * 256 + tid;
      int c = idx >> 5, kk = idx & 31;
      Bs[kk][c] = (_Float16)w_qkv[(size_t)(colBase + c) * K + kb + kk];
    }
    if (kb + 32 < K)  // prefetch next A chunk -> global_prefetch_b8
      __builtin_prefetch(&values[(size_t)(rowBase + (tid >> 1)) * K + kb + 32], 0, 1);
    __syncthreads();

    // A operand: wave's 16 rows, K=32.
    V16U a;
    const int ar = wave * 16 + l16;
    a.u[0] = *(const uint4*)&As[ar][half * 8];
    a.u[1] = *(const uint4*)&As[ar][16 + half * 8];
#pragma unroll
    for (int ti = 0; ti < 4; ++ti) {
      V16U b;
      for (int s = 0; s < 16; ++s)
        b.h[s] = Bs[half * 8 + (s & 7) + (s >> 3) * 16][ti * 16 + l16];
      acc[ti] = WMMA_F16(a.h, b.h, acc[ti]);
    }
    __syncthreads();
  }

  // Epilogue: bias + scatter to q / kt / v (f16).
#pragma unroll
  for (int ti = 0; ti < 4; ++ti) {
#pragma unroll
    for (int v = 0; v < 8; ++v) {
      int row = rowBase + wave * 16 + v + 8 * half;  // global point index
      int c   = colBase + ti * 16 + l16;             // 0..767
      float val = acc[ti][v] + b_qkv[c];
      int t = track_ids[row];
      int pos = row - t * L;
      int which = c >> 8, h = (c >> 5) & 7, dh = c & 31;
      _Float16 hv = (_Float16)val;
      size_t th = (size_t)(t * H + h);
      if (which == 0)      qbuf[(th * L + pos) * Dh + dh] = hv;
      else if (which == 1) ktbuf[(th * Dh + dh) * L + pos] = hv;  // transposed
      else                 vbuf[(th * L + pos) * Dh + dh] = hv;
    }
  }
}

// ---------------------------------------------------------------------------
// Kernel 2: flash attention per (track, head). Grid: T*H = 2048 blocks.
// Each of 8 waves owns 32 query rows; streams key/value columns 32 at a time.
// ---------------------------------------------------------------------------
__global__ __launch_bounds__(256) void attn_kernel(
    const _Float16* __restrict__ qbuf, const _Float16* __restrict__ ktbuf,
    const _Float16* __restrict__ vbuf, float* __restrict__ ctx) {
  constexpr int L = 256, Dh = 32, H = 8;
  __shared__ __align__(16) _Float16 Qs[256][40];     // [pos][dh]
  __shared__ __align__(16) _Float16 Ks[32][264];     // [dh][pos] (transposed)
  __shared__ __align__(16) _Float16 Vs[256][40];     // [pos][dh]
  __shared__ __align__(16) _Float16 Ps[8][32][40];   // per-wave P bounce

  const int tid  = threadIdx.x;
  const int lane = tid & 31, wave = tid >> 5;
  const int half = lane >> 4, l16 = lane & 15;
  const int t = blockIdx.x / H, h = blockIdx.x % H;

  const _Float16* qg = qbuf + (size_t)(t * H + h) * L * Dh;
  const _Float16* kg = ktbuf + (size_t)(t * H + h) * Dh * L;
  const _Float16* vg = vbuf + (size_t)(t * H + h) * L * Dh;

  // Stage Q, V (256x32) and Kt (32x256) in 16-byte chunks.
#if defined(HAVE_ASYNC_LDS)
  // CDNA5 async DMA: cache -> LDS without touching VGPRs; ASYNCcnt tracked.
#pragma unroll
  for (int e = 0; e < 4; ++e) {
    int c = e * 256 + tid;
    int pos = c >> 2, k8 = (c & 3) * 8;
    async_copy16(&qg[pos * Dh + k8], &Qs[pos][k8]);
    async_copy16(&vg[pos * Dh + k8], &Vs[pos][k8]);
    int dh = c >> 5, p8 = (c & 31) * 8;
    async_copy16(&kg[dh * L + p8], &Ks[dh][p8]);
  }
  wait_asynccnt0();
#else
#pragma unroll
  for (int e = 0; e < 4; ++e) {
    int c = e * 256 + tid;
    int pos = c >> 2, k8 = (c & 3) * 8;
    *(uint4*)&Qs[pos][k8] = *(const uint4*)&qg[pos * Dh + k8];
    *(uint4*)&Vs[pos][k8] = *(const uint4*)&vg[pos * Dh + k8];
  }
#pragma unroll
  for (int e = 0; e < 4; ++e) {
    int c = e * 256 + tid;
    int dh = c >> 5, p8 = (c & 31) * 8;
    *(uint4*)&Ks[dh][p8] = *(const uint4*)&kg[dh * L + p8];
  }
#endif
  __syncthreads();

  // Q A-operands (rows wave*32 + mt*16 + l16, K = Dh = 32).
  V16U aQ[2];
#pragma unroll
  for (int mt = 0; mt < 2; ++mt) {
    int r = wave * 32 + mt * 16 + l16;
    aQ[mt].u[0] = *(const uint4*)&Qs[r][half * 8];
    aQ[mt].u[1] = *(const uint4*)&Qs[r][16 + half * 8];
  }

  v8f acc[2][2];
  float mst[2][8], lst[2][8];
#pragma unroll
  for (int mt = 0; mt < 2; ++mt) {
#pragma unroll
    for (int dt = 0; dt < 2; ++dt) acc[mt][dt] = vzero8();
#pragma unroll
    for (int v = 0; v < 8; ++v) { mst[mt][v] = -1.0e30f; lst[mt][v] = 0.0f; }
  }

  const float rs = 0.17677669529663687f;  // 1/sqrt(32)

  for (int jb = 0; jb < 8; ++jb) {  // 32 key columns per step
    V16U bK0, bK1;
    for (int s = 0; s < 16; ++s) {
      int kk = half * 8 + (s & 7) + (s >> 3) * 16;
      bK0.h[s] = Ks[kk][jb * 32 + l16];
      bK1.h[s] = Ks[kk][jb * 32 + 16 + l16];
    }

#pragma unroll
    for (int mt = 0; mt < 2; ++mt) {
      v8f s0 = WMMA_F16(aQ[mt].h, bK0.h, vzero8());
      v8f s1 = WMMA_F16(aQ[mt].h, bK1.h, vzero8());
#pragma unroll
      for (int v = 0; v < 8; ++v) {
        float a0 = s0[v] * rs, a1 = s1[v] * rs;
        float rm = fmaxf(a0, a1);
#pragma unroll
        for (int off = 1; off < 16; off <<= 1)
          rm = fmaxf(rm, __shfl_xor(rm, off, 32));   // stays within half-wave
        float mprev = mst[mt][v];
        float mnew  = fmaxf(mprev, rm);
        float corr  = __expf(mprev - mnew);
        float p0 = __expf(a0 - mnew), p1 = __expf(a1 - mnew);
        float rsum = p0 + p1;
#pragma unroll
        for (int off = 1; off < 16; off <<= 1) rsum += __shfl_xor(rsum, off, 32);
        lst[mt][v] = lst[mt][v] * corr + rsum;
        mst[mt][v] = mnew;
#pragma unroll
        for (int dt = 0; dt < 2; ++dt) acc[mt][dt][v] *= corr;
        int pr = mt * 16 + v + 8 * half;  // D layout -> P row
        Ps[wave][pr][l16]      = (_Float16)p0;
        Ps[wave][pr][16 + l16] = (_Float16)p1;
      }
    }
    lds_wave_fence();  // wave-private P: in-order DS + compiler fence

    V16U bV0, bV1;
    for (int s = 0; s < 16; ++s) {
      int vr = jb * 32 + half * 8 + (s & 7) + (s >> 3) * 16;
      bV0.h[s] = Vs[vr][l16];
      bV1.h[s] = Vs[vr][16 + l16];
    }
#pragma unroll
    for (int mt = 0; mt < 2; ++mt) {
      V16U aP;
      int pr = mt * 16 + l16;
      aP.u[0] = *(const uint4*)&Ps[wave][pr][half * 8];
      aP.u[1] = *(const uint4*)&Ps[wave][pr][16 + half * 8];
      acc[mt][0] = WMMA_F16(aP.h, bV0.h, acc[mt][0]);
      acc[mt][1] = WMMA_F16(aP.h, bV1.h, acc[mt][1]);
    }
  }

  // Normalize and write ctx[t*L+pos][h*32+dh] (f32).
#pragma unroll
  for (int mt = 0; mt < 2; ++mt) {
#pragma unroll
    for (int dt = 0; dt < 2; ++dt) {
#pragma unroll
      for (int v = 0; v < 8; ++v) {
        int pos = wave * 32 + mt * 16 + v + 8 * half;
        float o = acc[mt][dt][v] / lst[mt][v];
        ctx[((size_t)(t * L + pos)) * 256 + h * 32 + dt * 16 + l16] = o;
      }
    }
  }
}

// ---------------------------------------------------------------------------
// Kernel 3: out = ctx @ w_lin^T + b_lin (f32 out). Grid: (512, 4).
// ---------------------------------------------------------------------------
__global__ __launch_bounds__(256) void out_gemm_kernel(
    const float* __restrict__ ctx, const float* __restrict__ w_lin,
    const float* __restrict__ b_lin, float* __restrict__ out) {
  constexpr int K = 256;
  __shared__ __align__(16) _Float16 As[128][40];
  __shared__ __align__(16) _Float16 Bs[32][72];

  const int tid  = threadIdx.x;
  const int lane = tid & 31, wave = tid >> 5;
  const int half = lane >> 4, l16 = lane & 15;
  const int rowBase = blockIdx.x * 128;
  const int colBase = blockIdx.y * 64;

  v8f acc[4];
#pragma unroll
  for (int ti = 0; ti < 4; ++ti) acc[ti] = vzero8();

  for (int kb = 0; kb < K; kb += 32) {
#pragma unroll
    for (int e = 0; e < 16; ++e) {
      int idx = e * 256 + tid;
      int r = idx >> 5, kk = idx & 31;
      As[r][kk] = (_Float16)ctx[(size_t)(rowBase + r) * K + kb + kk];
    }
#pragma unroll
    for (int e = 0; e < 8; ++e) {
      int idx = e * 256 + tid;
      int c = idx >> 5, kk = idx & 31;
      Bs[kk][c] = (_Float16)w_lin[(size_t)(colBase + c) * K + kb + kk];
    }
    if (kb + 32 < K)
      __builtin_prefetch(&ctx[(size_t)(rowBase + (tid >> 1)) * K + kb + 32], 0, 1);
    __syncthreads();

    V16U a;
    const int ar = wave * 16 + l16;
    a.u[0] = *(const uint4*)&As[ar][half * 8];
    a.u[1] = *(const uint4*)&As[ar][16 + half * 8];
#pragma unroll
    for (int ti = 0; ti < 4; ++ti) {
      V16U b;
      for (int s = 0; s < 16; ++s)
        b.h[s] = Bs[half * 8 + (s & 7) + (s >> 3) * 16][ti * 16 + l16];
      acc[ti] = WMMA_F16(a.h, b.h, acc[ti]);
    }
    __syncthreads();
  }

#pragma unroll
  for (int ti = 0; ti < 4; ++ti) {
#pragma unroll
    for (int v = 0; v < 8; ++v) {
      int row = rowBase + wave * 16 + v + 8 * half;
      int c   = colBase + ti * 16 + l16;
      out[(size_t)row * 256 + c] = acc[ti][v] + b_lin[c];
    }
  }
}

// ---------------------------------------------------------------------------
extern "C" void kernel_launch(void* const* d_in, const int* in_sizes, int n_in,
                              void* d_out, int out_size, void* d_ws, size_t ws_size,
                              hipStream_t stream) {
  const float* values   = (const float*)d_in[0];
  const float* w_qkv    = (const float*)d_in[1];
  const float* b_qkv    = (const float*)d_in[2];
  const float* w_lin    = (const float*)d_in[3];
  const float* b_lin    = (const float*)d_in[4];
  const int*   track_id = (const int*)d_in[5];
  (void)in_sizes; (void)n_in; (void)out_size; (void)ws_size;

  char* ws = (char*)d_ws;
  _Float16* qbuf  = (_Float16*)(ws);                           // 32 MB
  _Float16* ktbuf = (_Float16*)(ws + ((size_t)32 << 20));      // 32 MB
  _Float16* vbuf  = (_Float16*)(ws + ((size_t)64 << 20));      // 32 MB
  float*    ctx   = (float*)(ws + ((size_t)96 << 20));         // 64 MB

  qkv_gemm_kernel<<<dim3(512, 12), 256, 0, stream>>>(values, w_qkv, b_qkv,
                                                     track_id, qbuf, ktbuf, vbuf);
  attn_kernel<<<dim3(2048), 256, 0, stream>>>(qbuf, ktbuf, vbuf, ctx);
  out_gemm_kernel<<<dim3(512, 4), 256, 0, stream>>>(ctx, w_lin, b_lin,
                                                    (float*)d_out);
}
